// GraphLinear_90958817395177
// MI455X (gfx1250) — matrix operand order repaired and checked by reading, not compiled, not run on hardware
//
#include <hip/hip_runtime.h>

// GraphLinear fused kernel for MI455X (gfx1250, wave32, WMMA f16 16x16x32).
// out[b,n,o] = sum_m g[n,m] * (sum_i in[b,m,i] * W[type[m],o,i]) + bias[o]
// B=16384, N=128, DIN=DOUT=64, T=16.
// Fusion: per 16-row b-tile, h (128 x (16b*64o)) lives in LDS as pre-packed
// f16 WMMA B-operand vectors (256 KB), g packed as A-operand (32 KB in LDS).

typedef __attribute__((ext_vector_type(16))) _Float16 v16h;
typedef __attribute__((ext_vector_type(8)))  float    v8f;

#define NB_B     16384
#define NN       128
#define NDIN     64
#define NDOUT    64
#define NTYPES   16

// d_ws dword layout: [0,32768) wPack (128 KB), [32768,40960) gPack (32 KB)
#define WS_GOFF  32768
// LDS dword layout: [0,65536) Hs packed, [65536,73728) gPack copy
#define LDS_HDW  65536
#define LDS_GDW  8192
#define LDS_BYTES ((LDS_HDW + LDS_GDW) * 4)

static __device__ __forceinline__ unsigned int pk_f16(float a, float b) {
  // v_cvt_pk_rtz_f16_f32: two f32 -> packed f16 dword (low = a)
  auto h = __builtin_amdgcn_cvt_pkrtz(a, b);
  unsigned int u;
  __builtin_memcpy(&u, &h, 4);
  return u;
}

static __device__ __forceinline__ v16h ld16_u4(const uint4* p, size_t idx) {
  union { uint4 q[2]; v16h v; } u;
  u.q[0] = p[idx];
  u.q[1] = p[idx + 1];
  return u.v;
}

// ---------------------------------------------------------------------------
// Pre-pack: weights and g -> f16 WMMA operand layouts in d_ws.
// g as A-operand (16-bit, 16Mx32K): lane = m%16 + 16*(k>=16 half), dword v
// holds k = 2v,2v+1 within the half-range.
// W as B-operand (16-bit, 32Kx16N): lane = n%16 + 16*hi, VGPR0-3: K=kb+0..7,
// VGPR4-7: K=kb+16..23, kb = 8*hi.
// ---------------------------------------------------------------------------
__global__ void GraphLinear_prepack(const float* __restrict__ g,
                                    const float* __restrict__ w,
                                    unsigned int* __restrict__ ws) {
  const int bid = blockIdx.x;
  const int tid = threadIdx.x;
  if (bid < NTYPES) {
    // weight pack: vec id = kc*4+tc; value(kc,tc,l,v) = W[t][16tc+l%16][32kc+8*(l/16)+2v+(v>=4?8:0)(+1)]
    const int t  = bid;
    const int kc = tid >> 7;          // 0..1
    const int tc = (tid >> 5) & 3;    // 0..3
    const int l  = tid & 31;
    const int o  = tc * 16 + (l & 15);
    const int ib = kc * 32 + ((l >> 4) << 3);
    const float* wr = w + ((size_t)t * NDOUT + o) * NDIN + ib;
    unsigned int* dst = ws + (size_t)(((t * 8 + kc * 4 + tc) * 32 + l)) * 8;
#pragma unroll
    for (int v = 0; v < 8; ++v) dst[v] = pk_f16(wr[2 * v + (v >= 4 ? 8 : 0)], wr[2 * v + (v >= 4 ? 8 : 0) + 1]);
  } else {
    // g pack as A operand: tiles (nc 0..7, mc 0..3)
    const int j  = (bid - NTYPES) * 256 + tid;   // 0..1023
    const int nc = j >> 7;
    const int mc = (j >> 5) & 3;
    const int l  = j & 31;
    const int n  = nc * 16 + (l & 15);
    const int kb = (l >> 4) * 16;
    const float* gr = g + (size_t)n * NN + mc * 32;
    unsigned int* dst = ws + WS_GOFF + (size_t)(((nc * 4 + mc) * 32 + l)) * 8;
#pragma unroll
    for (int v = 0; v < 8; ++v) {
      const int mo = kb + 2 * v;
      dst[v] = pk_f16(gr[mo], gr[mo + 1]);
    }
  }
}

// Per-node linear: acc[tc] (16b x 16o f32) += in_tile(16x64) @ W[t]^T chunk.
static __device__ __forceinline__ void node_linear(
    const float* __restrict__ input, const unsigned int* __restrict__ ws,
    const int* __restrict__ ntype, int m, int b0, int lane, v8f acc[4]) {
  const int lm = lane & 15, lh = lane >> 4;
  const int kb = lh * 8;                       // A-operand K base for this lane
  const int t  = ntype[m];
  const float4* f =
      (const float4*)(input + ((size_t)(b0 + lm) * NN + m) * NDIN);
  const int q = kb >> 2;                       // 0 or 2 (float4 units)
  // runs: [kb,kb+8) [kb+16,kb+24) -> a0 ; +32 -> a1
  float4 f0 = f[q],     f1 = f[q + 1],  f2 = f[q + 4],  f3 = f[q + 5];
  float4 f4 = f[q + 8], f5 = f[q + 9],  f6 = f[q + 12], f7 = f[q + 13];
  union { unsigned int u[8]; v16h v; } a0, a1;
  a0.u[0] = pk_f16(f0.x, f0.y); a0.u[1] = pk_f16(f0.z, f0.w);
  a0.u[2] = pk_f16(f1.x, f1.y); a0.u[3] = pk_f16(f1.z, f1.w);
  a0.u[4] = pk_f16(f2.x, f2.y); a0.u[5] = pk_f16(f2.z, f2.w);
  a0.u[6] = pk_f16(f3.x, f3.y); a0.u[7] = pk_f16(f3.z, f3.w);
  a1.u[0] = pk_f16(f4.x, f4.y); a1.u[1] = pk_f16(f4.z, f4.w);
  a1.u[2] = pk_f16(f5.x, f5.y); a1.u[3] = pk_f16(f5.z, f5.w);
  a1.u[4] = pk_f16(f6.x, f6.y); a1.u[5] = pk_f16(f6.z, f6.w);
  a1.u[6] = pk_f16(f7.x, f7.y); a1.u[7] = pk_f16(f7.z, f7.w);
  const uint4* wq = (const uint4*)ws;
#pragma unroll
  for (int tc = 0; tc < 4; ++tc) {
    v16h b0v = ld16_u4(wq, (size_t)((t * 8 + tc) * 32 + lane) * 2);
    acc[tc] = __builtin_amdgcn_wmma_f32_16x16x32_f16(
        false, a0.v, false, b0v, (short)0, acc[tc], false, false);
    v16h b1v = ld16_u4(wq, (size_t)((t * 8 + 4 + tc) * 32 + lane) * 2);
    acc[tc] = __builtin_amdgcn_wmma_f32_16x16x32_f16(
        false, a1.v, false, b1v, (short)0, acc[tc], false, false);
  }
}

__global__ __launch_bounds__(256, 1) void GraphLinear_fused(
    const float* __restrict__ input, const float* __restrict__ bias,
    const int* __restrict__ ntype, const unsigned int* __restrict__ ws,
    float* __restrict__ out) {
  extern __shared__ unsigned int lds[];
  unsigned int* Hs = lds;             // packed B-operand vecs of h
  unsigned int* Gs = lds + LDS_HDW;   // packed A-operand vecs of g

  const int tid  = threadIdx.x;
  const int lane = tid & 31;
  const int wave = tid >> 5;          // 0..7
  const int lm = lane & 15, lh = lane >> 4;
  const int b0 = blockIdx.x * 16;

  // stage gPack into LDS (8192 dwords = 2048 uint4)
  {
    const uint4* src = (const uint4*)(ws + WS_GOFF);
    uint4* dst = (uint4*)Gs;
#pragma unroll
    for (int k = 0; k < 8; ++k) dst[tid + 256 * k] = src[tid + 256 * k];
  }

  // ---------------- Phase A: h[m] = in_tile @ W[type[m]]^T, m pairs --------
  for (int p = 0; p < 8; ++p) {
    const int m0 = wave * 16 + 2 * p;
    v8f acc0[4] = {v8f{}, v8f{}, v8f{}, v8f{}};
    v8f acc1[4] = {v8f{}, v8f{}, v8f{}, v8f{}};
    node_linear(input, ws, ntype, m0,     b0, lane, acc0);
    node_linear(input, ws, ntype, m0 + 1, b0, lane, acc1);
    // write into Hs as B-operand vectors; nodes m0/m0+1 share one dword
    const int mc  = m0 >> 5;                  // K-chunk of 32 m
    const int km  = m0 & 31;
    const int lp  = lm + (km & 16);           // dest lane slot
    const int vdw = (km & 15) >> 1;           // dword within vec
#pragma unroll
    for (int tc = 0; tc < 4; ++tc) {
#pragma unroll
      for (int r = 0; r < 8; ++r) {
        const int cc = (r + 8 * lh) * 4 + tc;          // 16-col chunk (b*4+tc)
        Hs[((mc * 64 + cc) * 32 + lp) * 8 + vdw] =
            pk_f16(acc0[tc][r], acc1[tc][r]);
      }
    }
  }
  __syncthreads();

  // ---------------- Phase B: OUT(128n x 1024c) = G @ H, + bias -------------
  const float biasv[4] = {bias[lm], bias[16 + lm], bias[32 + lm],
                          bias[48 + lm]};
  const uint4* Gq = (const uint4*)Gs;
  const uint4* Hq = (const uint4*)Hs;
  for (int nc = 0; nc < 8; ++nc) {
    v8f acc[8] = {v8f{}, v8f{}, v8f{}, v8f{}, v8f{}, v8f{}, v8f{}, v8f{}};
#pragma unroll
    for (int mc = 0; mc < 4; ++mc) {
      v16h av = ld16_u4(Gq, (size_t)(((nc * 4 + mc) * 32 + lane)) * 2);
#pragma unroll
      for (int cl = 0; cl < 8; ++cl) {
        v16h bv = ld16_u4(
            Hq, (size_t)(((mc * 64 + (wave * 8 + cl)) * 32 + lane)) * 2);
        acc[cl] = __builtin_amdgcn_wmma_f32_16x16x32_f16(
            false, av, false, bv, (short)0, acc[cl], false, false);
      }
    }
#pragma unroll
    for (int cl = 0; cl < 8; ++cl) {
      const int ccg = wave * 8 + cl;          // global 16-col chunk
      const int bl  = ccg >> 2;               // local b row
      const int o   = (ccg & 3) * 16 + lm;    // output feature
      const float bb = biasv[ccg & 3];
#pragma unroll
      for (int r = 0; r < 8; ++r) {
        const int n = nc * 16 + r + 8 * lh;
        out[((size_t)(b0 + bl) * NN + n) * NDOUT + o] = acc[cl][r] + bb;
      }
    }
  }
}

extern "C" void kernel_launch(void* const* d_in, const int* in_sizes, int n_in,
                              void* d_out, int out_size, void* d_ws,
                              size_t ws_size, hipStream_t stream) {
  (void)in_sizes; (void)n_in; (void)out_size; (void)ws_size;
  const float* input  = (const float*)d_in[0];
  const float* g      = (const float*)d_in[1];
  const float* weight = (const float*)d_in[2];
  const float* bias   = (const float*)d_in[3];
  const int*   ntype  = (const int*)d_in[4];
  float* out = (float*)d_out;
  unsigned int* ws = (unsigned int*)d_ws;   // needs 160 KB

  GraphLinear_prepack<<<NTYPES + 4, 256, 0, stream>>>(g, weight, ws);

  hipFuncSetAttribute(reinterpret_cast<const void*>(GraphLinear_fused),
                      hipFuncAttributeMaxDynamicSharedMemorySize, LDS_BYTES);
  GraphLinear_fused<<<NB_B / 16, 256, LDS_BYTES, stream>>>(input, bias, ntype,
                                                           ws, out);
}
